// ExpireSpanGPT_9844065042497
// MI455X (gfx1250) — matrix operand order repaired and checked by reading, not compiled, not run on hardware
//
#include <hip/hip_runtime.h>
#include <math.h>
#include <stdint.h>

// ---------------- CDNA5 types ----------------
typedef __attribute__((ext_vector_type(16))) __bf16 v16bf;
typedef __attribute__((ext_vector_type(8)))  float  v8f;

#define DEPTH 4
#define Dm    512
#define Hh    8
#define DHd   64
#define Nn    1024
#define Mm    1024
#define Vv    32000
#define Bb    2
#define CTX   (Mm + Nn)   // 2048

__device__ __forceinline__ unsigned short f2bf(float f) {
  unsigned u = __float_as_uint(f);
  u += 0x7FFFu + ((u >> 16) & 1u);   // round-to-nearest-even
  return (unsigned short)(u >> 16);
}

// async Global->LDS 16B copy (ASYNCcnt path, bypasses VGPRs)
__device__ __forceinline__ void async_b128(unsigned lds_off, const void* gsrc) {
  asm volatile("global_load_async_to_lds_b128 %0, %1, off"
               :: "v"(lds_off), "v"((unsigned long long)(uintptr_t)gsrc)
               : "memory");
}
__device__ __forceinline__ void wait_async0() {
  asm volatile("s_wait_asynccnt 0x0" ::: "memory");
}

// ---------------- generic bf16 WMMA GEMM ----------------
// C[M,N] = A[M,K] (bf16) * B (bf16, opt transposed) + bias, epilogues:
// EPI 0: none, 1: +res (residual), 2: exact GELU.
// Block 256 thr = 8 waves; tile 128x128x32; wave computes 32x64 (2x4 wmma).
// Double-buffered LDS; A and transposed-B staged with async-to-LDS.
template <int EPI, int TRANSB>
__global__ __launch_bounds__(256) void k_gemm(
    const unsigned short* __restrict__ A, const unsigned short* __restrict__ Bm,
    const float* __restrict__ bias, const float* __restrict__ res,
    float* __restrict__ Cf, unsigned short* __restrict__ Cb,
    int M, int N, int K, int lda, int ldb, int ldc, int batchH,
    long long sAb, long long sAh, long long sBb, long long sBh,
    long long sCb, long long sCh)
{
  // padded row stride 40 bf16 (= 5 uint4); 2 buffers each for A and B
  __shared__ unsigned short As[2 * 128 * 40];
  __shared__ unsigned short Bs[2 * 128 * 40];   // stored [n][k]
  uint4* As4 = (uint4*)As;
  uint4* Bs4 = (uint4*)Bs;

  const int tid = threadIdx.x;
  const int bz  = blockIdx.z;
  const int bb  = bz / batchH, hh = bz % batchH;
  A  += (size_t)(bb * sAb + hh * sAh);
  Bm += (size_t)(bb * sBb + hh * sBh);
  const long long coff = (long long)bb * sCb + (long long)hh * sCh;

  const int row0 = blockIdx.y * 128;
  const int col0 = blockIdx.x * 128;

  const int lane = tid & 31;
  const int wv   = tid >> 5;
  const int wm   = wv & 3;        // 4 waves along M (32 rows each)
  const int wn   = wv >> 2;       // 2 waves along N (64 cols each)
  const int half = lane >> 4;
  const int l16  = lane & 15;

  v8f acc[2][4] = {};

  // stage tile (k0) into LDS buffer sb
  auto stage = [&](int sb, int k0) {
    // ---- A tile 128x32: async b128 per lane ----
#pragma unroll
    for (int it = 0; it < 2; ++it) {
      int g  = tid + it * 256;          // 512 groups of 8 bf16
      int r  = g >> 2;
      int cg = (g & 3) * 8;
      const unsigned short* src = A + (size_t)(row0 + r) * lda + k0 + cg;
      unsigned dst = (unsigned)(uintptr_t)(&As4[sb * 640 + r * 5 + (g & 3)]);
      async_b128(dst, src);
    }
    // ---- B tile -> Bs[n][k] ----
    if (TRANSB) {
#pragma unroll
      for (int it = 0; it < 2; ++it) {
        int g  = tid + it * 256;
        int j  = g >> 2;
        int cg = (g & 3) * 8;
        if (col0 + j < N) {
          const unsigned short* src = Bm + (size_t)(col0 + j) * ldb + k0 + cg;
          unsigned dst = (unsigned)(uintptr_t)(&Bs4[sb * 640 + j * 5 + (g & 3)]);
          async_b128(dst, src);
        } else {
          uint4 z = {0u, 0u, 0u, 0u};
          Bs4[sb * 640 + j * 5 + (g & 3)] = z;
        }
      }
    } else {
      // layout transform: sync load + LDS scatter
#pragma unroll
      for (int it = 0; it < 2; ++it) {
        int g  = tid + it * 256;
        int c  = g >> 4;               // k within tile
        int jg = (g & 15) * 8;         // n group
        if (col0 + jg < N) {
          union { uint4 q; unsigned short s[8]; } t;
          t.q = *(const uint4*)(Bm + (size_t)(k0 + c) * ldb + col0 + jg);
#pragma unroll
          for (int u = 0; u < 8; ++u) Bs[sb * 5120 + (jg + u) * 40 + c] = t.s[u];
        } else {
#pragma unroll
          for (int u = 0; u < 8; ++u) Bs[sb * 5120 + (jg + u) * 40 + c] = 0;
        }
      }
      // prefetch next slab (global_prefetch_b8)
      if (k0 + 32 < K)
        __builtin_prefetch(Bm + (size_t)(k0 + 32 + (tid >> 7)) * ldb + col0 + (tid & 127), 0, 1);
    }
  };

  const int nk = K >> 5;
  stage(0, 0);
  wait_async0();
  __syncthreads();

  for (int kk = 0; kk < nk; ++kk) {
    const int buf = kk & 1;
    if (kk + 1 < nk) stage(buf ^ 1, (kk + 1) * 32);   // overlap with compute

    // ---- fragments per ISA 16-bit A(16x32)/B(32x16) layouts ----
    union Frag { uint4 u[2]; v16bf v; } afr[2], bfr[4];
#pragma unroll
    for (int fm = 0; fm < 2; ++fm) {
      int r = wm * 32 + fm * 16 + l16;
      afr[fm].u[0] = As4[buf * 640 + r * 5 + half];        // K = half*8 .. +7
      afr[fm].u[1] = As4[buf * 640 + r * 5 + 2 + half];    // K = 16+half*8 .. +7
    }
#pragma unroll
    for (int fn = 0; fn < 4; ++fn) {
      int nj = wn * 64 + fn * 16 + l16;
      bfr[fn].u[0] = Bs4[buf * 640 + nj * 5 + half];           // K = half*8 .. +7
      bfr[fn].u[1] = Bs4[buf * 640 + nj * 5 + 2 + half];       // K = 16+half*8 .. +7
    }
#pragma unroll
    for (int fm = 0; fm < 2; ++fm)
#pragma unroll
      for (int fn = 0; fn < 4; ++fn)
        acc[fm][fn] = __builtin_amdgcn_wmma_f32_16x16x32_bf16(
            false, afr[fm].v, false, bfr[fn].v, (short)0, acc[fm][fn], false, false);

    wait_async0();     // my async ops for the next stage have landed
    __syncthreads();   // everyone's landed; safe to reuse the other buffer
  }

  // ---- epilogue + store (C layout: elem e -> row half*8+e, col l16) ----
#pragma unroll
  for (int fm = 0; fm < 2; ++fm)
#pragma unroll
    for (int fn = 0; fn < 4; ++fn)
#pragma unroll
      for (int e = 0; e < 8; ++e) {
        int row = row0 + wm * 32 + fm * 16 + half * 8 + e;
        int col = col0 + wn * 64 + fn * 16 + l16;
        if (row < M && col < N) {
          float v = acc[fm][fn][e];
          if (bias) v += bias[col];
          size_t idx = (size_t)coff + (size_t)row * ldc + col;
          if (EPI == 1) v += res[idx];
          if (EPI == 2) v = 0.5f * v * (1.0f + erff(v * 0.70710678f));
          if (Cf) Cf[idx] = v;
          if (Cb) Cb[idx] = f2bf(v);
        }
      }
}

// ---------------- small kernels ----------------
__global__ void k_zero_aux(float* aux) { if (threadIdx.x < Bb) aux[threadIdx.x] = 0.0f; }

__global__ void k_cast(const float* __restrict__ s, unsigned short* __restrict__ d, size_t n) {
  size_t i = (size_t)blockIdx.x * 256 + threadIdx.x;
  if (i < n) d[i] = f2bf(s[i]);
}

__global__ void k_embed(const int* __restrict__ tok, const float* __restrict__ emb,
                        float* __restrict__ xf, unsigned short* __restrict__ xb) {
  size_t i = (size_t)blockIdx.x * 256 + threadIdx.x;   // over B*N*D
  int d = i & (Dm - 1);
  size_t bi = i >> 9;
  int t = tok[bi];
  float v = emb[(size_t)t * Dm + d];
  xf[i] = v;
  xb[i] = f2bf(v);
}

__global__ void k_pe(unsigned short* __restrict__ pe) {
  size_t i = (size_t)blockIdx.x * 256 + threadIdx.x;   // Nn*Dm
  int d   = i & (Dm - 1);
  int row = i >> 9;
  float t = (float)(Nn - 1 - row);
  int k   = (d < Dm / 2) ? d : d - Dm / 2;
  float inv = __expf(-((2.0f * k) / Dm) * logf(10000.0f));
  float ang = t * inv;
  pe[i] = f2bf((d < Dm / 2) ? __sinf(ang) : __cosf(ang));
}

__global__ void k_build_ctx(const float* __restrict__ mem, const float* __restrict__ xf,
                            unsigned short* __restrict__ ctx) {
  size_t i = (size_t)blockIdx.x * 256 + threadIdx.x;   // B*CTX*Dm
  int d = i & (Dm - 1);
  size_t rj = i >> 9;
  int j = rj & (CTX - 1);
  int b = rj >> 11;
  float v = (j < Mm) ? mem[((size_t)b * Mm + j) * Dm + d]
                     : xf[((size_t)b * Nn + (j - Mm)) * Dm + d];
  ctx[i] = f2bf(v);
}

// exps = sigmoid(mem@Wexp + bexp)*1024 ; em = clip((exps - t)/128 + 1, 0, 1);
// aux[b] += exps * 1[0<em<1] / (128*1024)
__global__ void k_expire(const float* __restrict__ mem, const int* __restrict__ times,
                         const float* __restrict__ wexp, const float* __restrict__ bexp,
                         float* __restrict__ em, float* __restrict__ aux) {
  __shared__ float red[256];
  int row = blockIdx.x;               // b*Mm + j
  int b = row >> 10;
  const float* mr = mem + (size_t)row * Dm;
  int t = threadIdx.x;
  red[t] = mr[t] * wexp[t] + mr[t + 256] * wexp[t + 256];
  __syncthreads();
  for (int s = 128; s > 0; s >>= 1) { if (t < s) red[t] += red[t + s]; __syncthreads(); }
  if (t == 0) {
    float dot = red[0] + bexp[0];
    float ex  = 1024.0f / (1.0f + __expf(-dot));
    float r   = ex - (float)times[row];
    float e   = fminf(fmaxf(r * (1.0f / 128.0f) + 1.0f, 0.0f), 1.0f);
    em[row] = e;
    if (e > 0.0f && e < 1.0f) atomicAdd(&aux[b], ex * (1.0f / (128.0f * 1024.0f)));
  }
}

// scale, TXL skew-gather of pd, causal mask, softmax, post-softmax expire gate
__global__ void k_softmax(const float* __restrict__ dots, const float* __restrict__ pd0,
                          const float* __restrict__ em, unsigned short* __restrict__ P) {
  __shared__ float red[256];
  int row = blockIdx.x;               // (b*H + h)*Nn + i
  int i   = row & (Nn - 1);
  int bh  = row >> 10;
  int b   = bh >> 3;
  int t   = threadIdx.x;
  size_t base = (size_t)row * CTX;
  const float sc = 0.125f;            // DH^-0.5
  float vals[8];
#pragma unroll
  for (int q = 0; q < 8; ++q) {
    int j = q * 256 + t;
    float v = dots[base + j] * sc;
    if (j >= Mm) {
      int jo = j - Mm;
      int idx = (i + 1) * Nn + jo;    // relative-shift trick on indices
      int r = idx / (Nn + 1);
      int c = idx % (Nn + 1);
      if (c > 0) v += pd0[((size_t)bh * Nn + r) * Nn + (c - 1)] * sc;
    }
    if (j > i + Mm) v = -1e10f;
    vals[q] = v;
  }
  float mx = vals[0];
#pragma unroll
  for (int q = 1; q < 8; ++q) mx = fmaxf(mx, vals[q]);
  red[t] = mx; __syncthreads();
  for (int s = 128; s > 0; s >>= 1) { if (t < s) red[t] = fmaxf(red[t], red[t + s]); __syncthreads(); }
  mx = red[0]; __syncthreads();
  float sum = 0.0f;
#pragma unroll
  for (int q = 0; q < 8; ++q) { vals[q] = __expf(vals[q] - mx); sum += vals[q]; }
  red[t] = sum; __syncthreads();
  for (int s = 128; s > 0; s >>= 1) { if (t < s) red[t] += red[t + s]; __syncthreads(); }
  float inv = 1.0f / red[0];
#pragma unroll
  for (int q = 0; q < 8; ++q) {
    int j = q * 256 + t;
    float g = (j < Mm) ? em[b * Mm + j] : 1.0f;
    P[base + j] = f2bf(vals[q] * inv * g);
  }
}

__global__ void k_ln(const float* __restrict__ x, const float* __restrict__ g,
                     const float* __restrict__ bl, unsigned short* __restrict__ out) {
  __shared__ float rs[256], rq[256];
  int row = blockIdx.x;               // B*Nn
  const float* xr = x + (size_t)row * Dm;
  int t = threadIdx.x;
  float a = xr[t], c = xr[t + 256];
  rs[t] = a + c; rq[t] = a * a + c * c;
  __syncthreads();
  for (int s = 128; s > 0; s >>= 1) { if (t < s) { rs[t] += rs[t + s]; rq[t] += rq[t + s]; } __syncthreads(); }
  float mu  = rs[0] * (1.0f / Dm);
  float var = rq[0] * (1.0f / Dm) - mu * mu;
  float inv = rsqrtf(var + 1e-5f);
  out[(size_t)row * Dm + t]       = f2bf((a - mu) * inv * g[t]       + bl[t]);
  out[(size_t)row * Dm + t + 256] = f2bf((c - mu) * inv * g[t + 256] + bl[t + 256]);
}

// ---------------- host-side GEMM dispatcher ----------------
static void gemm(hipStream_t st, int epi, bool transB,
                 const unsigned short* A, const unsigned short* B,
                 const float* bias, const float* res,
                 float* Cf, unsigned short* Cb,
                 int M, int N, int K, int lda, int ldb, int ldc,
                 int bB = 1, int bH = 1,
                 long long sAb = 0, long long sAh = 0,
                 long long sBb = 0, long long sBh = 0,
                 long long sCb = 0, long long sCh = 0) {
  dim3 gr((N + 127) / 128, (M + 127) / 128, bB * bH);
  dim3 bl(256);
#define GCASE(E, T) k_gemm<E, T><<<gr, bl, 0, st>>>(A, B, bias, res, Cf, Cb, M, N, K, lda, ldb, ldc, bH, sAb, sAh, sBb, sBh, sCb, sCh)
  if (!transB) { if (epi == 0) GCASE(0, 0); else if (epi == 1) GCASE(1, 0); else GCASE(2, 0); }
  else         { if (epi == 0) GCASE(0, 1); else if (epi == 1) GCASE(1, 1); else GCASE(2, 1); }
#undef GCASE
}

// ---------------- orchestration ----------------
extern "C" void kernel_launch(void* const* d_in, const int* in_sizes, int n_in,
                              void* d_out, int out_size, void* d_ws, size_t ws_size,
                              hipStream_t stream) {
  (void)in_sizes; (void)n_in; (void)out_size; (void)ws_size;
  const int*   tokens = (const int*)  d_in[0];
  const float* mems   = (const float*)d_in[1];
  const int*   times  = (const int*)  d_in[2];
  const float* tokemb = (const float*)d_in[3];
  const float* Wq     = (const float*)d_in[4];
  const float* bq     = (const float*)d_in[5];
  const float* Wkv    = (const float*)d_in[6];
  const float* bkv    = (const float*)d_in[7];
  const float* Wo     = (const float*)d_in[8];
  const float* bo     = (const float*)d_in[9];
  const float* Wpos   = (const float*)d_in[10];
  const float* bpos   = (const float*)d_in[11];
  const float* Wexp   = (const float*)d_in[12];
  const float* bexp   = (const float*)d_in[13];
  const float* W1     = (const float*)d_in[14];
  const float* b1     = (const float*)d_in[15];
  const float* W2     = (const float*)d_in[16];
  const float* b2     = (const float*)d_in[17];
  const float* lng    = (const float*)d_in[18];
  const float* lnb    = (const float*)d_in[19];
  const float* Whead  = (const float*)d_in[20];

  float* logits = (float*)d_out;
  float* aux    = logits + (size_t)Bb * Nn * Vv;

  // ---- carve workspace ----
  char* basep = (char*)d_ws;
  size_t off = 0;
  auto carve = [&](size_t bytes) -> void* {
    void* p = basep + off;
    off = (off + bytes + 255) & ~(size_t)255;
    return p;
  };
  typedef unsigned short u16;
  float* xf   = (float*)carve((size_t)Bb * Nn * Dm * 4);
  u16*   xb   = (u16*)  carve((size_t)Bb * Nn * Dm * 2);
  u16*   ctxb = (u16*)  carve((size_t)Bb * CTX * Dm * 2);
  u16*   qb   = (u16*)  carve((size_t)Bb * Nn * Dm * 2);
  u16*   kvb  = (u16*)  carve((size_t)Bb * CTX * 2 * Dm * 2);
  u16*   peb  = (u16*)  carve((size_t)Nn * Dm * 2);
  u16*   posb = (u16*)  carve((size_t)Nn * DHd * 2);
  float* dots = (float*)carve((size_t)Bb * Hh * Nn * CTX * 4);
  float* pd0  = (float*)carve((size_t)Bb * Hh * Nn * Nn * 4);
  u16*   Pb   = (u16*)  carve((size_t)Bb * Hh * Nn * CTX * 2);
  u16*   aob  = (u16*)  carve((size_t)Bb * Nn * Dm * 2);
  u16*   h1b  = (u16*)  carve((size_t)Bb * Nn * 4 * Dm * 2);
  float* em   = (float*)carve((size_t)Bb * Mm * 4);
  u16*   xlnb = (u16*)  carve((size_t)Bb * Nn * Dm * 2);
  u16*   wqb  = (u16*)  carve((size_t)DEPTH * Dm * Dm * 2);
  u16*   wkvb = (u16*)  carve((size_t)DEPTH * Dm * 2 * Dm * 2);
  u16*   wob  = (u16*)  carve((size_t)DEPTH * Dm * Dm * 2);
  u16*   wposb= (u16*)  carve((size_t)DEPTH * Dm * DHd * 2);
  u16*   w1b  = (u16*)  carve((size_t)DEPTH * Dm * 4 * Dm * 2);
  u16*   w2b  = (u16*)  carve((size_t)DEPTH * 4 * Dm * Dm * 2);
  u16*   whb  = (u16*)  carve((size_t)Dm * Vv * 2);

  k_zero_aux<<<1, 32, 0, stream>>>(aux);

  auto castN = [&](const float* s, u16* d, size_t n) {
    k_cast<<<(unsigned)((n + 255) / 256), 256, 0, stream>>>(s, d, n);
  };
  castN(Wq,   wqb,  (size_t)DEPTH * Dm * Dm);
  castN(Wkv,  wkvb, (size_t)DEPTH * Dm * 2 * Dm);
  castN(Wo,   wob,  (size_t)DEPTH * Dm * Dm);
  castN(Wpos, wposb,(size_t)DEPTH * Dm * DHd);
  castN(W1,   w1b,  (size_t)DEPTH * Dm * 4 * Dm);
  castN(W2,   w2b,  (size_t)DEPTH * 4 * Dm * Dm);
  castN(Whead,whb,  (size_t)Dm * Vv);

  k_pe<<<(Nn * Dm) / 256, 256, 0, stream>>>(peb);
  k_embed<<<(Bb * Nn * Dm) / 256, 256, 0, stream>>>(tokens, tokemb, xf, xb);

  for (int l = 0; l < DEPTH; ++l) {
    const float* mem_l = mems + (size_t)l * Bb * Mm * Dm;
    const int*   tim_l = times + (size_t)l * Bb * Mm;

    k_expire<<<Bb * Mm, 256, 0, stream>>>(mem_l, tim_l, Wexp + (size_t)l * Dm, bexp + l, em, aux);
    k_build_ctx<<<(Bb * CTX * Dm) / 256, 256, 0, stream>>>(mem_l, xf, ctxb);

    // q = x @ Wq + bq  -> bf16
    gemm(stream, 0, false, xb, wqb + (size_t)l * Dm * Dm, bq + (size_t)l * Dm,
         nullptr, nullptr, qb, Bb * Nn, Dm, Dm, Dm, Dm, Dm);
    // kv = ctx @ Wkv + bkv -> bf16
    gemm(stream, 0, false, ctxb, wkvb + (size_t)l * Dm * 2 * Dm, bkv + (size_t)l * 2 * Dm,
         nullptr, nullptr, kvb, Bb * CTX, 2 * Dm, Dm, Dm, 2 * Dm, 2 * Dm);
    // pos = pe @ Wpos + bpos -> bf16
    gemm(stream, 0, false, peb, wposb + (size_t)l * Dm * DHd, bpos + (size_t)l * DHd,
         nullptr, nullptr, posb, Nn, DHd, Dm, Dm, DHd, DHd);
    // dots[b,h] = q @ k^T  (batched over b,h)
    gemm(stream, 0, true, qb, kvb, nullptr, nullptr, dots, nullptr,
         Nn, CTX, DHd, Dm, 2 * Dm, CTX,
         Bb, Hh, (long long)Nn * Dm, DHd, (long long)CTX * 2 * Dm, DHd,
         (long long)Hh * Nn * CTX, (long long)Nn * CTX);
    // pd0[b,h] = q @ pos^T
    gemm(stream, 0, true, qb, posb, nullptr, nullptr, pd0, nullptr,
         Nn, Nn, DHd, Dm, DHd, Nn,
         Bb, Hh, (long long)Nn * Dm, DHd, 0, 0,
         (long long)Hh * Nn * Nn, (long long)Nn * Nn);

    k_softmax<<<Bb * Hh * Nn, 256, 0, stream>>>(dots, pd0, em, Pb);

    // attout[b,h] = P @ V -> bf16, scattered to [b, n, h*64..]
    gemm(stream, 0, false, Pb, kvb + Dm, nullptr, nullptr, nullptr, aob,
         Nn, DHd, CTX, CTX, 2 * Dm, Dm,
         Bb, Hh, (long long)Hh * Nn * CTX, (long long)Nn * CTX,
         (long long)CTX * 2 * Dm, DHd,
         (long long)Nn * Dm, DHd);
    // x = x + attout @ Wo + bo
    gemm(stream, 1, false, aob, wob + (size_t)l * Dm * Dm, bo + (size_t)l * Dm,
         xf, xf, xb, Bb * Nn, Dm, Dm, Dm, Dm, Dm);
    // h1 = gelu(x @ W1 + b1) -> bf16
    gemm(stream, 2, false, xb, w1b + (size_t)l * Dm * 4 * Dm, b1 + (size_t)l * 4 * Dm,
         nullptr, nullptr, h1b, Bb * Nn, 4 * Dm, Dm, Dm, 4 * Dm, 4 * Dm);
    // x = x + h1 @ W2 + b2
    gemm(stream, 1, false, h1b, w2b + (size_t)l * 4 * Dm * Dm, b2 + (size_t)l * Dm,
         xf, xf, xb, Bb * Nn, Dm, 4 * Dm, 4 * Dm, Dm, Dm);
  }

  k_ln<<<Bb * Nn, 256, 0, stream>>>(xf, lng, lnb, xlnb);
  // logits = ln(x) @ Whead
  gemm(stream, 0, false, xlnb, whb, nullptr, nullptr, logits, nullptr,
       Bb * Nn, Vv, Dm, Dm, Vv, Vv);
}
